// Decoder_62405874811315
// MI455X (gfx1250) — matrix operand order repaired and checked by reading, not compiled, not run on hardware
//
#include <hip/hip_runtime.h>

typedef __attribute__((ext_vector_type(16))) _Float16 v16h;
typedef __attribute__((ext_vector_type(8)))  _Float16 v8h;
typedef __attribute__((ext_vector_type(8)))  float    v8f;

namespace cfg {
constexpr int L = 6, B = 4, S = 2048, D = 512, H = 8, DK = 64, F = 2048;
constexpr int HD = H * DK;  // 512
}

#define DEVI __device__ __forceinline__

DEVI v8f wmma_f16(v16h a, v16h b, v8f c) {
  // D = A(16x32 f16) * B(32x16 f16) + C(16x16 f32)
  return __builtin_amdgcn_wmma_f32_16x16x32_f16(false, a, false, b, (short)0, c,
                                                false, false);
}

DEVI v16h combine8(v8h lo, v8h hi) {
  return __builtin_shufflevector(lo, hi, 0, 1, 2, 3, 4, 5, 6, 7, 8, 9, 10, 11,
                                 12, 13, 14, 15);
}

// A-fragment 16x32 f16 (ISA 7.12.2): lane m = lane&15, kb = (lane>>4)*8,
// 16 elems cover K = {kb..kb+7} U {kb+16..kb+23}.
DEVI v16h load_a_frag(const _Float16* __restrict__ base, int ld, int lane) {
  const int m = lane & 15, kb = (lane >> 4) * 8;
  const _Float16* p = base + (size_t)m * ld + kb;
  return combine8(*(const v8h*)p, *(const v8h*)(p + 16));
}

// ---------------------------------------------------------------- casts
__global__ void cast_f32_to_f16(const float* __restrict__ src,
                                _Float16* __restrict__ dst, long n) {
  const long i = (long)blockIdx.x * blockDim.x + threadIdx.x;
  if (i < n) dst[i] = (_Float16)src[i];
}

// src [K,N] f32 -> dst [N,K] f16  (K,N multiples of 32)
__global__ void transpose_cast(const float* __restrict__ src,
                               _Float16* __restrict__ dst, int K, int N) {
  __shared__ __align__(32) _Float16 tile[32][33];
  const int nt = blockIdx.x * 32, kt = blockIdx.y * 32;
  const int tx = threadIdx.x & 31, ty = threadIdx.x >> 5;  // 256 thr: ty 0..7
#pragma unroll
  for (int r = 0; r < 32; r += 8)
    tile[ty + r][tx] = (_Float16)src[(size_t)(kt + ty + r) * N + nt + tx];
  __syncthreads();
#pragma unroll
  for (int r = 0; r < 32; r += 8)
    dst[(size_t)(nt + ty + r) * K + kt + tx] = tile[tx][ty + r];
}

// V [B,S,H*DK] f16 -> Vt [B,H,DK,S] f16
__global__ void transpose_v(const _Float16* __restrict__ v,
                            _Float16* __restrict__ vt) {
  using namespace cfg;
  __shared__ __align__(32) _Float16 tile[32][33];
  const int bh = blockIdx.z, b = bh / H, h = bh % H;
  const int s0 = blockIdx.x * 32, d0 = blockIdx.y * 32;
  const int tx = threadIdx.x & 31, ty = threadIdx.x >> 5;
  const _Float16* src = v + (size_t)b * S * HD + h * DK;
#pragma unroll
  for (int r = 0; r < 32; r += 8)
    tile[ty + r][tx] = src[(size_t)(s0 + ty + r) * HD + d0 + tx];
  __syncthreads();
  _Float16* dst = vt + ((size_t)b * H + h) * (size_t)DK * S;
#pragma unroll
  for (int r = 0; r < 32; r += 8)
    dst[(size_t)(d0 + ty + r) * S + s0 + tx] = tile[tx][ty + r];
}

// ---------------------------------------------------------------- GEMM
// C[M,N] = act(A[M,K] @ Bt[N,K]^T + bias[N]); one wave per block computes a
// 32x64 tile = 8 WMMA per k-step of 32. K is a compile-time constant (512 or
// 2048) so every fragment load is a single per-lane base + immediate offset
// (t*16*K*2 bytes <= 192 KB fits the 24-bit ioffset) -- no per-subtile 64-bit
// address registers, no spills. The compiler clause-groups the 6 loads per
// step and issues progressive s_wait_loadcnt across the 8 WMMAs.
template <bool RELU, int K>
__global__ void gemm32_wmma(const _Float16* __restrict__ A,
                            const _Float16* __restrict__ Bt,
                            const float* __restrict__ bias,
                            float* __restrict__ C32, _Float16* __restrict__ C16,
                            int N) {
  const int lane = threadIdx.x;
  const int nl = lane & 15;
  const int kg = lane >> 4;
  const int n0 = blockIdx.x * 64;
  const int m0 = blockIdx.y * 32;

  // per-lane base pointers; all k-loop addressing is immediate offsets
  const _Float16* ap = A + (size_t)(m0 + nl) * K + kg * 8;    // A row (m0+nl)
  const _Float16* bp = Bt + (size_t)(n0 + nl) * K + kg * 8;  // Bt row (n0+nl)

  v8f acc[2][4] = {};
  for (int k0 = 0; k0 < K; k0 += 32) {
    __builtin_prefetch(ap + k0 + 256, 0, 1);
    const v16h a0 = combine8(*(const v8h*)(ap + k0),
                             *(const v8h*)(ap + k0 + 16));
    const v16h a1 = combine8(*(const v8h*)(ap + 16 * K + k0),
                             *(const v8h*)(ap + 16 * K + k0 + 16));
    v16h b[4];
#pragma unroll
    for (int t = 0; t < 4; ++t)
      b[t] = combine8(*(const v8h*)(bp + (size_t)t * 16 * K + k0),
                      *(const v8h*)(bp + (size_t)t * 16 * K + k0 + 16));
#pragma unroll
    for (int t = 0; t < 4; ++t) {
      acc[0][t] = wmma_f16(a0, b[t], acc[0][t]);
      acc[1][t] = wmma_f16(a1, b[t], acc[1][t]);
    }
  }

#pragma unroll
  for (int mt = 0; mt < 2; ++mt)
#pragma unroll
    for (int t = 0; t < 4; ++t) {
      const int n = n0 + t * 16 + nl;
      const float bv = bias ? bias[n] : 0.0f;
#pragma unroll
      for (int r = 0; r < 8; ++r) {
        float v = acc[mt][t][r] + bv;
        if (RELU) v = fmaxf(v, 0.0f);
        const size_t idx = (size_t)(m0 + mt * 16 + kg * 8 + r) * N + n;
        if (C32) C32[idx] = v;
        if (C16) C16[idx] = (_Float16)v;
      }
    }
}

// ---------------------------------------------------------------- attention
// Flash-style: one wave handles 16 query rows of one (b,h); K-chunks of 32.
// V-chunk fragments and the NEXT K-chunk fragments are issued right after the
// score WMMAs so the online-softmax VALU work hides their latency.
template <bool CAUSAL>
__global__ void attn_wmma(const _Float16* __restrict__ Q,   // [B,Sq,HD]
                          const _Float16* __restrict__ Km,  // [B,Sk,HD]
                          const _Float16* __restrict__ Vt,  // [B,H,DK,Sk]
                          _Float16* __restrict__ O,         // [B,Sq,HD]
                          int Sq, int Sk) {
  using namespace cfg;
  const int lane = threadIdx.x;
  const int nl = lane & 15;
  const int kg = lane >> 4;
  const int q0 = blockIdx.x * 16;
  const int h = blockIdx.y;
  const int b = blockIdx.z;

  const _Float16* qp = Q + ((size_t)b * Sq + q0) * HD + h * DK;
  const _Float16* kp = Km + (size_t)b * Sk * HD + h * DK;
  const _Float16* vp = Vt + ((size_t)b * H + h) * (size_t)DK * Sk;

  const v16h qa0 = load_a_frag(qp, HD, lane);       // dk 0..31
  const v16h qa1 = load_a_frag(qp + 32, HD, lane);  // dk 32..63

  v8f oacc[4] = {};
  float rm[8], rl[8];
#pragma unroll
  for (int r = 0; r < 8; ++r) { rm[r] = -1e30f; rl[r] = 0.0f; }

  __shared__ __align__(32) _Float16 plds[16][32];

  // K^T B-fragments: B-frag cols are K rows, so per-lane loads are contiguous
  v16h kbf[2][2];
  auto loadK = [&](int s0c) {
#pragma unroll
    for (int t = 0; t < 2; ++t) {
      const _Float16* kbp = kp + (size_t)(s0c + t * 16 + nl) * HD + kg * 8;
      kbf[t][0] = combine8(*(const v8h*)kbp, *(const v8h*)(kbp + 16));
      kbf[t][1] = combine8(*(const v8h*)(kbp + 32), *(const v8h*)(kbp + 48));
    }
  };

  const int sEnd = CAUSAL ? (q0 + 16) : Sk;
  loadK(0);
  for (int s0 = 0; s0 < sEnd; s0 += 32) {
    // scores 16x32 = Q(16x64) @ K^T chunk
    v8f sc[2] = {};
#pragma unroll
    for (int t = 0; t < 2; ++t) {
      sc[t] = wmma_f16(qa0, kbf[t][0], sc[t]);
      sc[t] = wmma_f16(qa1, kbf[t][1], sc[t]);
    }
    // issue V chunk (consumed after softmax) and next K chunk now
    v16h vbr[4];
#pragma unroll
    for (int t = 0; t < 4; ++t)
      vbr[t] = combine8(*(const v8h*)(vp + (size_t)(t * 16 + nl) * Sk + s0 + kg * 8),
                        *(const v8h*)(vp + (size_t)(t * 16 + nl) * Sk + s0 + kg * 8 + 16));
    if (s0 + 32 < sEnd) loadK(s0 + 32);

    // scale (1/sqrt(64)) + causal mask in C-frag layout
#pragma unroll
    for (int t = 0; t < 2; ++t) {
      const int ng = s0 + t * 16 + nl;
#pragma unroll
      for (int r = 0; r < 8; ++r) {
        float v = sc[t][r] * 0.125f;
        if (CAUSAL && ng > q0 + kg * 8 + r) v = -1e30f;
        sc[t][r] = v;
      }
    }
    // online softmax; each half-wave's 16 lanes hold the 16 cols of 8 rows
#pragma unroll
    for (int r = 0; r < 8; ++r) {
      float mx = fmaxf(sc[0][r], sc[1][r]);
#pragma unroll
      for (int off = 8; off; off >>= 1) mx = fmaxf(mx, __shfl_xor(mx, off, 32));
      const float mnew = fmaxf(rm[r], mx);
      const float corr = __expf(rm[r] - mnew);
      const float e0 = __expf(sc[0][r] - mnew);
      const float e1 = __expf(sc[1][r] - mnew);
      sc[0][r] = e0;
      sc[1][r] = e1;
      float sm = e0 + e1;
#pragma unroll
      for (int off = 8; off; off >>= 1) sm += __shfl_xor(sm, off, 32);
      rl[r] = rl[r] * corr + sm;
      rm[r] = mnew;
#pragma unroll
      for (int t = 0; t < 4; ++t) oacc[t][r] *= corr;
    }
    // P: C-frag f32 -> LDS f16 -> A-frag layout for the PV WMMA
#pragma unroll
    for (int t = 0; t < 2; ++t)
#pragma unroll
      for (int r = 0; r < 8; ++r)
        plds[kg * 8 + r][t * 16 + nl] = (_Float16)sc[t][r];
    __syncthreads();  // single-wave WG -> S_NOP, acts as compiler fence
    const int kb2 = kg * 8;
    const v16h pa = combine8(*(const v8h*)&plds[nl][kb2],
                             *(const v8h*)&plds[nl][kb2 + 16]);
    __syncthreads();
    // O(16x64) += P(16x32) @ Vchunk(32x64)
#pragma unroll
    for (int t = 0; t < 4; ++t) oacc[t] = wmma_f16(pa, vbr[t], oacc[t]);
  }

  _Float16* op = O + ((size_t)b * Sq + q0) * HD + h * DK;
#pragma unroll
  for (int r = 0; r < 8; ++r) {
    const float inv = 1.0f / rl[r];
#pragma unroll
    for (int t = 0; t < 4; ++t)
      op[(size_t)(kg * 8 + r) * HD + t * 16 + nl] = (_Float16)(oacc[t][r] * inv);
  }
}

// ---------------------------------------------------------------- LN
// out = LN(resid + x) * gamma + beta ; 256 threads per row, D = 512.
__global__ void add_layernorm(const float* __restrict__ resid,
                              const float* __restrict__ x,
                              const float* __restrict__ gamma,
                              const float* __restrict__ beta,
                              float* __restrict__ out32,
                              _Float16* __restrict__ out16) {
  using namespace cfg;
  const int row = blockIdx.x;
  const int tid = threadIdx.x;
  const float* pr = resid + (size_t)row * D;
  const float* px = x + (size_t)row * D;
  const float v0 = pr[tid] + px[tid];
  const float v1 = pr[tid + 256] + px[tid + 256];
  float s = v0 + v1, q = v0 * v0 + v1 * v1;
#pragma unroll
  for (int off = 16; off; off >>= 1) {
    s += __shfl_xor(s, off, 32);
    q += __shfl_xor(q, off, 32);
  }
  __shared__ float sb[8], qb[8];
  const int wv = tid >> 5, ln = tid & 31;
  if (ln == 0) { sb[wv] = s; qb[wv] = q; }
  __syncthreads();
  if (tid == 0) {
    float ts = 0.f, tq = 0.f;
#pragma unroll
    for (int i = 0; i < 8; ++i) { ts += sb[i]; tq += qb[i]; }
    sb[0] = ts * (1.0f / D);
    qb[0] = tq * (1.0f / D);
  }
  __syncthreads();
  const float mean = sb[0];
  const float rstd = rsqrtf(qb[0] - mean * mean + 1e-5f);
  const float o0 = (v0 - mean) * rstd * gamma[tid] + beta[tid];
  const float o1 = (v1 - mean) * rstd * gamma[tid + 256] + beta[tid + 256];
  if (out32) {
    out32[(size_t)row * D + tid] = o0;
    out32[(size_t)row * D + tid + 256] = o1;
  }
  if (out16) {
    out16[(size_t)row * D + tid] = (_Float16)o0;
    out16[(size_t)row * D + tid + 256] = (_Float16)o1;
  }
}

// ---------------------------------------------------------------- host
extern "C" void kernel_launch(void* const* d_in, const int* in_sizes, int n_in,
                              void* d_out, int out_size, void* d_ws,
                              size_t ws_size, hipStream_t stream) {
  using namespace cfg;
  (void)in_sizes; (void)n_in; (void)out_size; (void)ws_size;
  const float* x = (const float*)d_in[0];
  const float* enc = (const float*)d_in[1];
  const float* saW[4] = {(const float*)d_in[2], (const float*)d_in[4],
                         (const float*)d_in[6], (const float*)d_in[8]};
  const float* saB[4] = {(const float*)d_in[3], (const float*)d_in[5],
                         (const float*)d_in[7], (const float*)d_in[9]};
  const float* caW[4] = {(const float*)d_in[10], (const float*)d_in[12],
                         (const float*)d_in[14], (const float*)d_in[16]};
  const float* caB[4] = {(const float*)d_in[11], (const float*)d_in[13],
                         (const float*)d_in[15], (const float*)d_in[17]};
  const float* lnG[3] = {(const float*)d_in[18], (const float*)d_in[20],
                         (const float*)d_in[22]};
  const float* lnB[3] = {(const float*)d_in[19], (const float*)d_in[21],
                         (const float*)d_in[23]};
  const float* w1 = (const float*)d_in[24];
  const float* b1 = (const float*)d_in[25];
  const float* w2 = (const float*)d_in[26];
  const float* b2 = (const float*)d_in[27];

  char* wp = (char*)d_ws;
  auto alloc = [&](size_t bytes) -> void* {
    void* r = (void*)wp;
    wp += (bytes + 255) & ~(size_t)255;
    return r;
  };
  const size_t BS = (size_t)B * S;
  _Float16* enc16 = (_Float16*)alloc(BS * D * 2);
  _Float16* h16 = (_Float16*)alloc(BS * D * 2);
  float* h32b = (float*)alloc(BS * D * 4);
  float* x1_32 = (float*)alloc(BS * D * 4);
  _Float16* x1_16 = (_Float16*)alloc(BS * D * 2);
  _Float16* y16 = (_Float16*)alloc(BS * D * 2);
  _Float16* q16 = (_Float16*)alloc(BS * HD * 2);
  _Float16* k16 = (_Float16*)alloc(BS * HD * 2);
  _Float16* v16 = (_Float16*)alloc(BS * HD * 2);
  _Float16* vt16 = (_Float16*)alloc(BS * HD * 2);
  _Float16* att16 = (_Float16*)alloc(BS * HD * 2);
  float* tmp32 = (float*)alloc(BS * D * 4);
  _Float16* f116 = (_Float16*)alloc(BS * (size_t)F * 2);
  _Float16* wtSA[4];
  _Float16* wtCA[4];
  for (int i = 0; i < 4; ++i) wtSA[i] = (_Float16*)alloc((size_t)D * HD * 2);
  for (int i = 0; i < 4; ++i) wtCA[i] = (_Float16*)alloc((size_t)D * HD * 2);
  _Float16* w1t = (_Float16*)alloc((size_t)D * F * 2);
  _Float16* w2t = (_Float16*)alloc((size_t)F * D * 2);

  {
    const long n = (long)BS * D;
    cast_f32_to_f16<<<dim3((unsigned)((n + 255) / 256)), dim3(256), 0, stream>>>(
        enc, enc16, n);
    cast_f32_to_f16<<<dim3((unsigned)((n + 255) / 256)), dim3(256), 0, stream>>>(
        x, h16, n);
  }

  // K is compile-time: 512 for all projections, 2048 (=F) for the second FFN
  auto gemm = [&](const _Float16* A, const _Float16* Bt, const float* bias,
                  float* c32, _Float16* c16, int M, int N, int K, bool relu) {
    dim3 grid(N / 64, M / 32);
    if (K == 512) {
      if (relu)
        gemm32_wmma<true, 512><<<grid, dim3(32), 0, stream>>>(A, Bt, bias, c32,
                                                              c16, N);
      else
        gemm32_wmma<false, 512><<<grid, dim3(32), 0, stream>>>(A, Bt, bias, c32,
                                                               c16, N);
    } else {
      if (relu)
        gemm32_wmma<true, 2048><<<grid, dim3(32), 0, stream>>>(A, Bt, bias, c32,
                                                               c16, N);
      else
        gemm32_wmma<false, 2048><<<grid, dim3(32), 0, stream>>>(A, Bt, bias,
                                                                c32, c16, N);
    }
  };

  const float* h32 = x;
  for (int l = 0; l < L; ++l) {
    // weights -> f16, transposed to [N][K] so B-fragments load contiguously
    for (int i = 0; i < 4; ++i) {
      const int K0 = (i < 3) ? D : HD;  // wq/wk/wv: [D,HD]; wo: [HD,D]
      const int N0 = (i < 3) ? HD : D;
      transpose_cast<<<dim3(N0 / 32, K0 / 32), dim3(256), 0, stream>>>(
          saW[i] + (size_t)l * K0 * N0, wtSA[i], K0, N0);
      transpose_cast<<<dim3(N0 / 32, K0 / 32), dim3(256), 0, stream>>>(
          caW[i] + (size_t)l * K0 * N0, wtCA[i], K0, N0);
    }
    transpose_cast<<<dim3(F / 32, D / 32), dim3(256), 0, stream>>>(
        w1 + (size_t)l * D * F, w1t, D, F);
    transpose_cast<<<dim3(D / 32, F / 32), dim3(256), 0, stream>>>(
        w2 + (size_t)l * F * D, w2t, F, D);

    // ---- masked self-attention: x1 = LN(h + SA(h)) ----
    gemm(h16, wtSA[0], saB[0] + (size_t)l * HD, nullptr, q16, (int)BS, HD, D, false);
    gemm(h16, wtSA[1], saB[1] + (size_t)l * HD, nullptr, k16, (int)BS, HD, D, false);
    gemm(h16, wtSA[2], saB[2] + (size_t)l * HD, nullptr, v16, (int)BS, HD, D, false);
    transpose_v<<<dim3(S / 32, DK / 32, B * H), dim3(256), 0, stream>>>(v16, vt16);
    attn_wmma<true><<<dim3(S / 16, H, B), dim3(32), 0, stream>>>(q16, k16, vt16,
                                                                 att16, S, S);
    gemm(att16, wtSA[3], saB[3] + (size_t)l * D, tmp32, nullptr, (int)BS, D, HD, false);
    add_layernorm<<<dim3((unsigned)BS), dim3(256), 0, stream>>>(
        h32, tmp32, lnG[0] + (size_t)l * D, lnB[0] + (size_t)l * D, x1_32, x1_16);

    // ---- cross-attention: y = LN(h + CA(x1, enc))  (residual from h) ----
    gemm(x1_16, wtCA[0], caB[0] + (size_t)l * HD, nullptr, q16, (int)BS, HD, D, false);
    gemm(enc16, wtCA[1], caB[1] + (size_t)l * HD, nullptr, k16, (int)BS, HD, D, false);
    gemm(enc16, wtCA[2], caB[2] + (size_t)l * HD, nullptr, v16, (int)BS, HD, D, false);
    transpose_v<<<dim3(S / 32, DK / 32, B * H), dim3(256), 0, stream>>>(v16, vt16);
    attn_wmma<false><<<dim3(S / 16, H, B), dim3(32), 0, stream>>>(q16, k16, vt16,
                                                                  att16, S, S);
    gemm(att16, wtCA[3], caB[3] + (size_t)l * D, tmp32, nullptr, (int)BS, D, HD, false);
    add_layernorm<<<dim3((unsigned)BS), dim3(256), 0, stream>>>(
        h32, tmp32, lnG[1] + (size_t)l * D, lnB[1] + (size_t)l * D, nullptr, y16);

    // ---- FFN: h' = LN(x1 + relu(y@w1+b1)@w2+b2)  (residual from x1) ----
    gemm(y16, w1t, b1 + (size_t)l * F, nullptr, f116, (int)BS, F, D, true);
    gemm(f116, w2t, b2 + (size_t)l * D, tmp32, nullptr, (int)BS, D, F, false);
    float* hn = (l == L - 1) ? (float*)d_out : h32b;
    add_layernorm<<<dim3((unsigned)BS), dim3(256), 0, stream>>>(
        x1_32, tmp32, lnG[2] + (size_t)l * D, lnB[2] + (size_t)l * D, hn, h16);
    h32 = hn;
  }
}
